// kernel_generated_0_62904091017382
// MI455X (gfx1250) — matrix operand order repaired and checked by reading, not compiled, not run on hardware
//
#include <hip/hip_runtime.h>
#include <hip/hip_bf16.h>

typedef _Float16 half8  __attribute__((ext_vector_type(8)));
typedef _Float16 half16 __attribute__((ext_vector_type(16)));
typedef float    floatx8 __attribute__((ext_vector_type(8)));

#define B_    256
#define H_    56
#define W_    56
#define CO_   64
#define RH    4              // h-rows per workgroup
#define SLOTS 9              // 6 shared S-slices + 3 special (h==0 wrap)
#define SROWF 296            // f16 per w-row: 9*32=288 + 8 pad -> 592B (16B mult)

// ---------------------------------------------------------------------------
// Kernel 1: rearrange weights (32,3,3,64) fp32 -> wT f16 [k][i][kk = j*32+g]
// ---------------------------------------------------------------------------
__global__ void k_prep_weights_62904091017382(const float* __restrict__ w,
                                              _Float16* __restrict__ wt) {
    int e = blockIdx.x * blockDim.x + threadIdx.x;
    if (e >= 3 * CO_ * 96) return;
    int kk = e % 96;
    int gi = e / 96;
    int i  = gi % CO_;
    int k  = gi / CO_;
    int g  = kk & 31;
    int j  = kk >> 5;
    wt[e] = (_Float16)w[((g * 3 + j) * 3 + k) * CO_ + i];
}

// ---------------------------------------------------------------------------
// Kernel 2: one WG per (b, h-quad). 256 threads = 8 waves.
// Uses s[h][j] == S[h+j] (h>=1) to share the im2col build across 4 h-rows,
// and keeps all 18 B fragments in registers across the h-loop.
// ---------------------------------------------------------------------------
__global__ void __launch_bounds__(256)
k_conv_62904091017382(const float* __restrict__ x,
                      const _Float16* __restrict__ wt,
                      float* __restrict__ out) {
    __shared__ __align__(16) _Float16 s_lds[57 * SROWF];   // [w-row 0..55, 56=zeros][slot][g]
    __shared__ __align__(16) float    o_lds[CO_ * 57];     // [i][o], stride 57 floats

    const int wg  = blockIdx.x;
    const int b   = wg / (H_ / RH);
    const int h0  = (wg - b * (H_ / RH)) * RH;
    const int tid = threadIdx.x;

    // ---- zero row 56 across all slots ----
    for (int idx = tid; idx < SROWF; idx += 256)
        s_lds[56 * SROWF + idx] = (_Float16)0.f;

    // ---- shared S-slices: slot s (u = h0+s, s=0..5):
    //      S[u][g][w] = x[g, u-1, w] + x[32+g, u-2, w]  (zero OOB) ----
    for (int e = tid; e < 6 * 32 * W_; e += 256) {
        int s  = e / (32 * W_);
        int r  = e - s * (32 * W_);
        int g  = r / W_;
        int ww = r - g * W_;
        int u  = h0 + s;
        int r0 = u - 1, r1 = u - 2;
        float v = 0.f;
        if ((unsigned)r0 < (unsigned)H_)
            v += x[(((size_t)b * 64 + g) * H_ + r0) * W_ + ww];
        if ((unsigned)r1 < (unsigned)H_)
            v += x[(((size_t)b * 64 + 32 + g) * H_ + r1) * W_ + ww];
        s_lds[ww * SROWF + s * 32 + g] = (_Float16)v;
    }
    // ---- special slices for the h==0 wrap (slots 6..8), first quad only ----
    if (h0 == 0) {
        for (int e = tid; e < 3 * 32 * W_; e += 256) {
            int j  = e / (32 * W_);
            int r  = e - j * (32 * W_);
            int g  = r / W_;
            int ww = r - g * W_;
            int r0 = j - 1;        // ch g
            int r1 = 54 + j;       // ch 32+g (wrapped), invalid for j==2
            float v = 0.f;
            if ((unsigned)r0 < (unsigned)H_)
                v += x[(((size_t)b * 64 + g) * H_ + r0) * W_ + ww];
            if (r1 < H_)
                v += x[(((size_t)b * 64 + 32 + g) * H_ + r1) * W_ + ww];
            s_lds[ww * SROWF + (6 + j) * 32 + g] = (_Float16)v;
        }
    }
    __syncthreads();

    // ---- per-wave tile assignment ----
    const int wave   = tid >> 5;
    const int lane   = tid & 31;
    const int laneM  = lane & 15;
    const int laneHi = lane >> 4;
    const int m_base = (wave & 3) * 16;        // o-tile
    const int npair  = wave >> 2;              // i tiles {2n, 2n+1}
    const int iA     = npair * 32 + laneM;
    const int iB     = iA + 16;

    // A-row byte offsets per k-shift (independent of hh): row_w(o,k) in s_lds
    int aoff[3];
    {
        int o = m_base + laneM;
        #pragma unroll
        for (int k = 0; k < 3; ++k) {
            int q = o + k - 1;                       // valid tap iff 0<=q<=55
            int row;
            if ((unsigned)q < (unsigned)W_) row = (q == 0) ? (W_ - 1) : (q - 1);
            else                            row = 56;
            aoff[k] = row * (SROWF * 2) + laneHi * 16;   // bytes
        }
    }

    union HU { half16 v; half8 h[2]; };

    // ---- preload ALL B fragments (reused across the 4 h-rows) ----
    HU Bf[3][3][2];
    const char* wb = (const char*)wt + (size_t)iA * 192 + laneHi * 16;
    #pragma unroll
    for (int k = 0; k < 3; ++k) {
        #pragma unroll
        for (int t = 0; t < 3; ++t) {
            const char* p = wb + k * (CO_ * 192) + 64 * t;
            Bf[k][t][0].h[0] = *(const half8*)(p);
            Bf[k][t][0].h[1] = *(const half8*)(p + 32);
            Bf[k][t][1].h[0] = *(const half8*)(p + 16 * 192);
            Bf[k][t][1].h[1] = *(const half8*)(p + 16 * 192 + 32);
        }
    }

    const char* sbytes = (const char*)s_lds;

    for (int hh = 0; hh < RH; ++hh) {
        const int ub = (h0 == 0 && hh == 0) ? 6 : hh;   // slot base for this h-row

        floatx8 c0 = {0.f, 0.f, 0.f, 0.f, 0.f, 0.f, 0.f, 0.f};
        floatx8 c1 = c0;

        #pragma unroll
        for (int k = 0; k < 3; ++k) {
            const char* arow = sbytes + aoff[k] + ub * 64;
            #pragma unroll
            for (int t = 0; t < 3; ++t) {
                HU A;
                A.h[0] = *(const half8*)(arow + 64 * t);
                A.h[1] = *(const half8*)(arow + 64 * t + 32);
                c0 = __builtin_amdgcn_wmma_f32_16x16x32_f16(false, A.v, false, Bf[k][t][0].v,
                                                            (short)0, c0, false, false);
                c1 = __builtin_amdgcn_wmma_f32_16x16x32_f16(false, A.v, false, Bf[k][t][1].v,
                                                            (short)0, c1, false, false);
            }
        }

        // C tiles -> o_lds[i][o] (C/D layout: VGPR v -> M = v + 8*laneHi, N = laneM)
        #pragma unroll
        for (int v = 0; v < 8; ++v) {
            int o = m_base + v + laneHi * 8;
            if (o < W_) {
                o_lds[iA * 57 + o] = c0[v];
                o_lds[iB * 57 + o] = c1[v];
            }
        }
        __syncthreads();

        // coalesced store: out[b, i, h0+hh, o]
        const size_t obase = (((size_t)b * CO_) * H_ + (h0 + hh)) * W_;
        for (int e = tid; e < CO_ * W_; e += 256) {
            int i = e / W_;
            int o = e - i * W_;
            out[obase + (size_t)i * (H_ * W_) + o] = o_lds[i * 57 + o];
        }
        __syncthreads();
    }
}

extern "C" void kernel_launch(void* const* d_in, const int* in_sizes, int n_in,
                              void* d_out, int out_size, void* d_ws, size_t ws_size,
                              hipStream_t stream) {
    const float* x = (const float*)d_in[0];   // (256, 64, 56, 56) fp32
    const float* w = (const float*)d_in[1];   // (32, 3, 3, 64)    fp32
    float* out     = (float*)d_out;           // (256, 64, 56, 56) fp32
    _Float16* wt   = (_Float16*)d_ws;         // 3*64*96 f16 = 36,864 B

    {
        int n = 3 * CO_ * 96;
        k_prep_weights_62904091017382<<<(n + 255) / 256, 256, 0, stream>>>(w, wt);
    }
    k_conv_62904091017382<<<B_ * (H_ / RH), 256, 0, stream>>>(x, wt, out);
}
